// RootModel_816043786337
// MI455X (gfx1250) — matrix operand order repaired and checked by reading, not compiled, not run on hardware
//
#include <hip/hip_runtime.h>
#include <hip/hip_bf16.h>
#include <cstdint>
#include <cstddef>

#define DFEAT 64
#define LAYERS 2
#define BN_EPS 1e-5f

typedef __attribute__((ext_vector_type(16))) _Float16 v16h;
typedef __attribute__((ext_vector_type(8)))  _Float16 v8h;
typedef __attribute__((ext_vector_type(8)))  float    v8f;

// ---------------- convert x (f32 -> f16) ----------------
__global__ void k_cvt_x(const float* __restrict__ x, _Float16* __restrict__ xh, int n) {
    int i = blockIdx.x * blockDim.x + threadIdx.x;
    int stride = gridDim.x * blockDim.x;
    for (; i < n; i += stride) xh[i] = (_Float16)x[i];
}

// ---------------- build WMMA B-fragments for all layers & matrices ----------------
// Fragment order (per ISA 16-bit B layout, K=32, N=16):
//   lane l: n = (l&15) + ntile*16 ; halves i=0..15 hold K = (l>>4)*8 + (i&7) + (i>>3)*16 + kstep*32
// Flat index: ((((layer*4+mat)*4+ntile)*2+kstep)*32+lane)*16 + i
__global__ void k_cvt_w(const float* __restrict__ Wk, const float* __restrict__ Wq,
                        const float* __restrict__ Wv, const float* __restrict__ Ws,
                        _Float16* __restrict__ wfrag) {
    int idx = blockIdx.x * blockDim.x + threadIdx.x;
    const int total = LAYERS * 4 * 4 * 2 * 32 * 16;   // 32768
    if (idx >= total) return;
    int i     =  idx        & 15;
    int lane  = (idx >> 4)  & 31;
    int kstep = (idx >> 9)  & 1;
    int ntile = (idx >> 10) & 3;
    int mat   = (idx >> 12) & 3;
    int l     = (idx >> 14);
    int k = ((lane >> 4) << 3) + (i & 7) + ((i >> 3) << 4) + kstep * 32;
    int n = (lane & 15) + ntile * 16;
    const float* W = (mat == 0) ? Wk : (mat == 1) ? Wq : (mat == 2) ? Wv : Ws;
    wfrag[idx] = (_Float16)W[((size_t)l * DFEAT + k) * DFEAT + n];
}

// ---------------- zero ----------------
__global__ void k_zero(float* __restrict__ p, int n) {
    int i = blockIdx.x * blockDim.x + threadIdx.x;
    int stride = gridDim.x * blockDim.x;
    for (; i < n; i += stride) p[i] = 0.0f;
}

// ---------------- fused 4-way GEMM: {K,Q,V,S} = xh @ {Wk,Wq,Wv,Ws} + b ----------------
// One wave per 16-node tile; 32 x v_wmma_f32_16x16x32_f16 per wave.
// B fragments for one matrix (8 x v16h = 64 VGPRs) are loaded as a batch so a
// single wait covers 8 consecutive WMMAs instead of one wait per WMMA.
__global__ void __launch_bounds__(256)
k_gemm(const _Float16* __restrict__ xh, const _Float16* __restrict__ wfrag,
       const float* __restrict__ bk, const float* __restrict__ bq,
       const float* __restrict__ bv, const float* __restrict__ bs,
       float* __restrict__ Ko, float* __restrict__ Qo,
       float* __restrict__ Vo, float* __restrict__ So, int ntiles) {
    int wave = (blockIdx.x * blockDim.x + threadIdx.x) >> 5;
    if (wave >= ntiles) return;                 // wave-uniform: EXEC all-ones inside
    int lane = threadIdx.x & 31;
    int hi = lane >> 4, lo = lane & 15;

    // A fragments (ISA 16-bit A 16x32 layout): two contiguous 16B chunks per lane
    v16h a[2];
#pragma unroll
    for (int ks = 0; ks < 2; ++ks) {
        const _Float16* p = xh + ((size_t)wave * 16 + lo) * DFEAT + hi * 8 + ks * 32;
        v8h alo = *(const v8h*)p;          // K = base .. base+7
        v8h ahi = *(const v8h*)(p + 16);   // K = base+16 .. base+23
        a[ks] = __builtin_shufflevector(alo, ahi,
                    0,1,2,3,4,5,6,7,8,9,10,11,12,13,14,15);
    }

    v8f c[16] = {};
#pragma unroll
    for (int mat = 0; mat < 4; ++mat) {
        v16h b[8];
#pragma unroll
        for (int nt = 0; nt < 4; ++nt)
#pragma unroll
            for (int ks = 0; ks < 2; ++ks)
                b[nt * 2 + ks] = *(const v16h*)(wfrag +
                    ((size_t)((mat * 4 + nt) * 2 + ks) * 32 + lane) * 16);
#pragma unroll
        for (int nt = 0; nt < 4; ++nt)
#pragma unroll
            for (int ks = 0; ks < 2; ++ks)
                c[mat * 4 + nt] = __builtin_amdgcn_wmma_f32_16x16x32_f16(
                    false, a[ks], false, b[nt * 2 + ks], (short)0,
                    c[mat * 4 + nt], false, false);
    }

    const float* bias[4] = {bk, bq, bv, bs};
    float* outp[4] = {Ko, Qo, Vo, So};
#pragma unroll
    for (int mat = 0; mat < 4; ++mat) {
#pragma unroll
        for (int nt = 0; nt < 4; ++nt) {
            float bb = bias[mat][nt * 16 + lo];
#pragma unroll
            for (int r = 0; r < 8; ++r) {   // C layout: M = r + 8*hi, N = lo
                size_t row = (size_t)wave * 16 + r + 8 * hi;
                outp[mat][row * DFEAT + nt * 16 + lo] = c[mat * 4 + nt][r] + bb;
            }
        }
    }
}

// ---------------- edge phase: agg[dst] += sigmoid(k[dst]+q[src]) * v[src] ----------------
// One wave per edge, float2 per lane (256B coalesced row reads); 2 edges in
// flight per iteration for memory-level parallelism against L2 latency.
__global__ void k_edge(const int* __restrict__ ei, int E,
                       const float* __restrict__ K, const float* __restrict__ Q,
                       const float* __restrict__ V, float* __restrict__ agg) {
    int wave = (blockIdx.x * blockDim.x + threadIdx.x) >> 5;
    int lane = threadIdx.x & 31;
    int nw = (gridDim.x * blockDim.x) >> 5;
    int fo = lane * 2;
    for (int e = wave; e < E; e += 2 * nw) {
        int e2 = e + nw;
        bool has2 = e2 < E;                 // wave-uniform
        int s0 = ei[e], d0 = ei[E + e];
        const float2 k0 = *(const float2*)(K + (size_t)d0 * DFEAT + fo);
        const float2 q0 = *(const float2*)(Q + (size_t)s0 * DFEAT + fo);
        const float2 v0 = *(const float2*)(V + (size_t)s0 * DFEAT + fo);
        int s1 = 0, d1 = 0;
        float2 k1, q1, v1;
        if (has2) {
            s1 = ei[e2]; d1 = ei[E + e2];
            k1 = *(const float2*)(K + (size_t)d1 * DFEAT + fo);
            q1 = *(const float2*)(Q + (size_t)s1 * DFEAT + fo);
            v1 = *(const float2*)(V + (size_t)s1 * DFEAT + fo);
        }
        float g0 = 1.0f / (1.0f + __expf(-(k0.x + q0.x)));
        float g1 = 1.0f / (1.0f + __expf(-(k0.y + q0.y)));
        float* ap0 = agg + (size_t)d0 * DFEAT + fo;
        atomicAdd(ap0,     g0 * v0.x);
        atomicAdd(ap0 + 1, g1 * v0.y);
        if (has2) {
            float h0 = 1.0f / (1.0f + __expf(-(k1.x + q1.x)));
            float h1 = 1.0f / (1.0f + __expf(-(k1.y + q1.y)));
            float* ap1 = agg + (size_t)d1 * DFEAT + fo;
            atomicAdd(ap1,     h0 * v1.x);
            atomicAdd(ap1 + 1, h1 * v1.y);
        }
    }
}

// ---------------- post: y = relu(agg + S + bias); per-feature sum/sumsq -> stats ----------------
__global__ void __launch_bounds__(256)
k_post(const float* __restrict__ agg, const float* __restrict__ S,
       const float* __restrict__ bias, float* __restrict__ y,
       float* __restrict__ stats, int total) {
    __shared__ float s_sum[DFEAT], s_sq[DFEAT];
    int tid = threadIdx.x;
    if (tid < DFEAT) { s_sum[tid] = 0.0f; s_sq[tid] = 0.0f; }
    __syncthreads();
    int i0 = blockIdx.x * blockDim.x + tid;
    int stride = gridDim.x * blockDim.x;        // multiple of 64 -> f fixed per thread
    int f = i0 & (DFEAT - 1);
    float bb = bias[f];
    float asum = 0.0f, asq = 0.0f;
    for (int i = i0; i < total; i += stride) {
        float v = agg[i] + S[i] + bb;
        v = v > 0.0f ? v : 0.0f;
        y[i] = v;
        asum += v; asq += v * v;
    }
    atomicAdd(&s_sum[f], asum);
    atomicAdd(&s_sq[f], asq);
    __syncthreads();
    if (tid < DFEAT)           atomicAdd(&stats[tid], s_sum[tid]);
    else if (tid < 2 * DFEAT)  atomicAdd(&stats[tid], s_sq[tid - DFEAT]);
}

// ---------------- BN apply: out = (y-mean)*rsqrt(var+eps)*gamma + beta ----------------
__global__ void __launch_bounds__(256)
k_bn(const float* __restrict__ y, const float* __restrict__ stats,
     const float* __restrict__ gamma, const float* __restrict__ beta,
     float* __restrict__ outf, _Float16* __restrict__ outh, int total, float invN) {
    int i0 = blockIdx.x * blockDim.x + threadIdx.x;
    int stride = gridDim.x * blockDim.x;        // multiple of 64
    int f = i0 & (DFEAT - 1);
    float mean = stats[f] * invN;
    float var = stats[DFEAT + f] * invN - mean * mean;
    float sc = rsqrtf(var + BN_EPS) * gamma[f];
    float sh = beta[f] - mean * sc;
    for (int i = i0; i < total; i += stride) {
        float v = y[i] * sc + sh;
        if (outf) outf[i] = v;
        if (outh) outh[i] = (_Float16)v;
    }
}

extern "C" void kernel_launch(void* const* d_in, const int* in_sizes, int n_in,
                              void* d_out, int out_size, void* d_ws, size_t ws_size,
                              hipStream_t stream) {
    const float* x     = (const float*)d_in[0];
    const int*   ei    = (const int*)  d_in[1];
    // d_in[2] (batch) unused: reference BN normalizes over the full node axis
    const float* Wk    = (const float*)d_in[3];
    const float* bk    = (const float*)d_in[4];
    const float* Wq    = (const float*)d_in[5];
    const float* bq    = (const float*)d_in[6];
    const float* Wv    = (const float*)d_in[7];
    const float* bv    = (const float*)d_in[8];
    const float* Ws    = (const float*)d_in[9];
    const float* bs    = (const float*)d_in[10];
    const float* bias  = (const float*)d_in[11];
    const float* gamma = (const float*)d_in[12];
    const float* beta  = (const float*)d_in[13];

    const int ND = in_sizes[0];          // N * 64
    const int N  = ND / DFEAT;
    const int E  = in_sizes[1] / 2;
    float* out = (float*)d_out;

    uint8_t* ws = (uint8_t*)d_ws;
    size_t off = 0;
    auto take = [&](size_t bytes) -> uint8_t* {
        uint8_t* p = ws + off;
        off = (off + bytes + 255) & ~(size_t)255;
        return p;
    };
    _Float16* xh    = (_Float16*)take((size_t)ND * 2);
    float*    Kb    = (float*)   take((size_t)ND * 4);
    float*    Qb    = (float*)   take((size_t)ND * 4);
    float*    Vb    = (float*)   take((size_t)ND * 4);
    float*    Sb    = (float*)   take((size_t)ND * 4);
    float*    agg   = (float*)   take((size_t)ND * 4);
    _Float16* wfrag = (_Float16*)take((size_t)LAYERS * 4 * DFEAT * DFEAT * 2);
    float*    stats = (float*)   take(2 * DFEAT * 4);
    float*    y     = Kb;   // K dead after edge phase; reuse as pre-BN activations

    k_cvt_x<<<1024, 256, 0, stream>>>(x, xh, ND);
    k_cvt_w<<<(LAYERS * 4 * 4 * 2 * 32 * 16 + 255) / 256, 256, 0, stream>>>(Wk, Wq, Wv, Ws, wfrag);

    const int ntiles = N / 16;                       // N = 100000 = 16 * 6250
    const int gemm_blocks = (ntiles * 32 + 255) / 256;

    for (int l = 0; l < LAYERS; ++l) {
        k_zero<<<1024, 256, 0, stream>>>(agg, ND);
        k_zero<<<1, 128, 0, stream>>>(stats, 2 * DFEAT);
        k_gemm<<<gemm_blocks, 256, 0, stream>>>(
            xh, wfrag + (size_t)l * 4 * DFEAT * DFEAT,
            bk + l * DFEAT, bq + l * DFEAT, bv + l * DFEAT, bs + l * DFEAT,
            Kb, Qb, Vb, Sb, ntiles);
        k_edge<<<2048, 256, 0, stream>>>(ei, E, Kb, Qb, Vb, agg);
        k_post<<<512, 256, 0, stream>>>(agg, Sb, bias + l * DFEAT, y, stats, ND);
        k_bn<<<512, 256, 0, stream>>>(y, stats, gamma + l * DFEAT, beta + l * DFEAT,
                                      (l == LAYERS - 1) ? out : nullptr,
                                      (l == LAYERS - 1) ? nullptr : xh,
                                      ND, 1.0f / (float)N);
    }
}
